// Upsampler_46428596469898
// MI455X (gfx1250) — matrix operand order repaired and checked by reading, not compiled, not run on hardware
//
#include <hip/hip_runtime.h>

#define BB 16
#define SS 512
#define CIN 256
#define HH 128
#define G3 384
#define TT 2048

#if defined(__has_builtin)
#if __has_builtin(__builtin_amdgcn_global_load_async_to_lds_b128) && \
    __has_builtin(__builtin_amdgcn_s_wait_asynccnt)
#define USE_ASYNC_LDS 1
#endif
#endif

typedef __attribute__((ext_vector_type(16))) __bf16 v16bf;
typedef __attribute__((ext_vector_type(8)))  float  v8f;

#if USE_ASYNC_LDS
// Pointer types matching the async-to-LDS builtin signature:
//   (int4 addrspace(1)*, int4 addrspace(3)*, imm offset, imm cpol)
typedef int v4i_ __attribute__((vector_size(4 * sizeof(int))));
typedef __attribute__((address_space(1))) v4i_* gasync_p;
typedef __attribute__((address_space(3))) v4i_* lasync_p;
#endif

// ---- WMMA fragment loaders (bf16 operands from f32 row-major memory) ----
// A: 16x32 (MxK). src points at tile origin (row m0, col k0), lda in elements.
__device__ __forceinline__ v16bf load_a_frag_g(const float* __restrict__ src,
                                               int lda, int l16, int half) {
  const float* p = src + (size_t)l16 * lda + 8 * half;
  v16bf a;
#pragma unroll
  for (int e = 0; e < 8; ++e) a[e] = (__bf16)p[e];
#pragma unroll
  for (int e = 0; e < 8; ++e) a[8 + e] = (__bf16)p[16 + e];
  return a;
}
// B: 32x16 (KxN) where memory is row-major [N][K] (i.e. B[k][n] = wrow[n*ldb+k]).
__device__ __forceinline__ v16bf load_b_fragT(const float* __restrict__ wrow,
                                              int ldb, int l16, int half) {
  const float* p = wrow + (size_t)l16 * ldb + 8 * half;
  v16bf b;
#pragma unroll
  for (int e = 0; e < 8; ++e) b[e] = (__bf16)p[e];
#pragma unroll
  for (int e = 0; e < 8; ++e) b[8 + e] = (__bf16)p[16 + e];
  return b;
}

// ---- Kernel 1: xp[d] = x @ W_ih[d]^T + b_ih[d]  (M=8192, N=384, K=256, d=0,1)
// Each wave computes 16x48 (3 N-tiles) so one A-fragment feeds 3 WMMAs.
__global__ __launch_bounds__(256) void k_xp_gemm(const float* __restrict__ x,
                                                 const float* __restrict__ Wih,
                                                 const float* __restrict__ bih,
                                                 float* __restrict__ xp) {
  int wid = blockIdx.x * 8 + (threadIdx.x >> 5);  // 8192 waves
  int lane = threadIdx.x & 31, l16 = lane & 15, half = lane >> 4;
  int ng = wid & 7;            // group of 3 N-tiles
  int mt = (wid >> 3) & 511;
  int dir = wid >> 12;
  int m0 = mt * 16;
  const float* W = Wih + (size_t)dir * G3 * CIN;
  v8f c[3] = {};
  for (int ks = 0; ks < CIN; ks += 32) {
    v16bf a = load_a_frag_g(x + (size_t)m0 * CIN + ks, CIN, l16, half);
#pragma unroll
    for (int j = 0; j < 3; ++j) {
      int n0 = (ng * 3 + j) * 16;
      v16bf bb = load_b_fragT(W + (size_t)n0 * CIN + ks, CIN, l16, half);
      c[j] = __builtin_amdgcn_wmma_f32_16x16x32_bf16(false, a, false, bb,
                                                     (short)0, c[j], false, false);
    }
  }
  float* out = xp + (size_t)dir * BB * SS * G3;
#pragma unroll
  for (int j = 0; j < 3; ++j) {
    int n0 = (ng * 3 + j) * 16;
    float bias = bih[dir * G3 + n0 + l16];
#pragma unroll
    for (int i = 0; i < 8; ++i) {
      int row = m0 + 8 * half + i;
      out[(size_t)row * G3 + n0 + l16] = c[j][i] + bias;
    }
  }
}

// ---- Kernel 2: sequential GRU scan, one block per direction ----
// Per step: gh = h @ W_hh^T + b_hh (M=16,N=384,K=128) via 96 WMMAs (12 per wave) spread
// over 8 waves; wave w owns N-tiles {w, w+8, w+16} so all three gates for
// h-columns [16w,16w+16) are wave-local. W_hh fragments persist in VGPRs.
// Gate inputs xp[:,t,:] are double-buffered in LDS via async-to-LDS loads
// (ASYNCcnt), issued one step ahead and overlapped with the WMMA chain.
__global__ __launch_bounds__(256) void k_gru_scan(const float* __restrict__ xp,
                                                  const float* __restrict__ Whh,
                                                  const float* __restrict__ bhh,
                                                  float* __restrict__ xout) {
  const int dir = blockIdx.x;
  const float* xpd = xp + (size_t)dir * BB * SS * G3;
  const float* W = Whh + (size_t)dir * G3 * HH;
  const float* bh = bhh + dir * G3;
  __shared__ float h[16 * 132];  // padded rows -> conflict-free A-frag reads
  int tid = threadIdx.x, wv = tid >> 5, lane = tid & 31;
  int l16 = lane & 15, half = lane >> 4;
  for (int i = tid; i < 16 * 132; i += 256) h[i] = 0.f;
  v16bf bf[3][4];  // 3 gate tiles x 4 k-steps, resident in registers
#pragma unroll
  for (int j = 0; j < 3; ++j) {
    int n0 = 16 * (wv + 8 * j);
#pragma unroll
    for (int k = 0; k < 4; ++k)
      bf[j][k] = load_b_fragT(W + (size_t)n0 * HH + 32 * k, HH, l16, half);
  }
  int hcol = 16 * wv + l16;
  float bhr = bh[hcol], bhz = bh[HH + hcol], bhn = bh[2 * HH + hcol];

#if USE_ASYNC_LDS
  extern __shared__ float gq[];  // [2][16*G3] double-buffered gate inputs
  int rowc[6], colc[6];          // 6 b128 chunks per thread cover 16x384 f32
#pragma unroll
  for (int cidx = 0; cidx < 6; ++cidx) {
    int i = tid + 256 * cidx;    // float4 index; 96 float4 per row
    rowc[cidx] = i / 96;
    colc[cidx] = (i % 96) * 4;
  }
  {  // prologue: fill buffer 0 with gi for the first timestep
    int t0 = dir ? (SS - 1) : 0;
#pragma unroll
    for (int cidx = 0; cidx < 6; ++cidx) {
      const float* gsrc = xpd + ((size_t)rowc[cidx] * SS + t0) * G3 + colc[cidx];
      float* ldst = gq + rowc[cidx] * G3 + colc[cidx];
      __builtin_amdgcn_global_load_async_to_lds_b128(
          (gasync_p)(void*)gsrc, (lasync_p)(void*)ldst, 0, 0);
    }
  }
  int cur = 0;
#endif
  __syncthreads();

  for (int step = 0; step < SS; ++step) {
    int t = dir ? (SS - 1 - step) : step;
    // A fragments from h in LDS
    v16bf a[4];
#pragma unroll
    for (int k = 0; k < 4; ++k) {
      const float* p = h + l16 * 132 + 32 * k + 8 * half;
#pragma unroll
      for (int e = 0; e < 8; ++e) a[k][e] = (__bf16)p[e];
#pragma unroll
      for (int e = 0; e < 8; ++e) a[k][8 + e] = (__bf16)p[16 + e];
    }
    __syncthreads();  // all h reads done before this step's h writes

#if USE_ASYNC_LDS
    if (step + 1 < SS) {  // kick off next step's gate inputs (overlaps WMMAs)
      int tn = dir ? (SS - 2 - step) : (step + 1);
      float* bufn = gq + (cur ^ 1) * (16 * G3);
#pragma unroll
      for (int cidx = 0; cidx < 6; ++cidx) {
        const float* gsrc = xpd + ((size_t)rowc[cidx] * SS + tn) * G3 + colc[cidx];
        float* ldst = bufn + rowc[cidx] * G3 + colc[cidx];
        __builtin_amdgcn_global_load_async_to_lds_b128(
            (gasync_p)(void*)gsrc, (lasync_p)(void*)ldst, 0, 0);
      }
    }
#else
    if (step + 1 < SS && tid < 16) {
      int tn = dir ? (SS - 2 - step) : step + 1;
      __builtin_prefetch(xpd + ((size_t)tid * SS + tn) * G3, 0, 1);
    }
#endif

    v8f cr = {}, cz = {}, cn = {};
#pragma unroll
    for (int k = 0; k < 4; ++k) {
      cr = __builtin_amdgcn_wmma_f32_16x16x32_bf16(false, a[k], false, bf[0][k],
                                                   (short)0, cr, false, false);
      cz = __builtin_amdgcn_wmma_f32_16x16x32_bf16(false, a[k], false, bf[1][k],
                                                   (short)0, cz, false, false);
      cn = __builtin_amdgcn_wmma_f32_16x16x32_bf16(false, a[k], false, bf[2][k],
                                                   (short)0, cn, false, false);
    }

#if USE_ASYNC_LDS
    // The 6 ops issued last step are older than the 6 just issued (in-order
    // completion) -> asynccnt<=6 means the current buffer is resident.
    if (step + 1 < SS) __builtin_amdgcn_s_wait_asynccnt(6);
    else               __builtin_amdgcn_s_wait_asynccnt(0);
    __syncthreads();  // current gi buffer complete across all waves
    const float* gib = gq + cur * (16 * G3);
#endif

    float hnew[8];
#pragma unroll
    for (int i = 0; i < 8; ++i) {
      int b = 8 * half + i;  // batch row per C-layout
#if USE_ASYNC_LDS
      float gir = gib[b * G3 + hcol];
      float giz = gib[b * G3 + HH + hcol];
      float gin = gib[b * G3 + 2 * HH + hcol];
#else
      const float* gi = xpd + ((size_t)b * SS + t) * G3;
      float gir = gi[hcol], giz = gi[HH + hcol], gin = gi[2 * HH + hcol];
#endif
      float ghr = cr[i] + bhr, ghz = cz[i] + bhz, ghn = cn[i] + bhn;
      float r = 1.f / (1.f + __expf(-(gir + ghr)));
      float z = 1.f / (1.f + __expf(-(giz + ghz)));
      float nn = tanhf(gin + r * ghn);
      hnew[i] = (1.f - z) * nn + z * h[b * 132 + hcol];
    }
#pragma unroll
    for (int i = 0; i < 8; ++i) {
      int b = 8 * half + i;
      h[b * 132 + hcol] = hnew[i];
      xout[((size_t)b * SS + t) * (2 * HH) + dir * HH + hcol] = hnew[i];
    }
#if USE_ASYNC_LDS
    cur ^= 1;
#endif
    __syncthreads();  // h writes visible before next step's A-build
  }
}

// ---- Kernel 3: projection head + logsumexp + factor + dur cumsum -> centers/range
__global__ __launch_bounds__(256) void k_head(const float* __restrict__ x2,
                                              const float* __restrict__ projw,
                                              const float* __restrict__ projb,
                                              const float* __restrict__ mask,
                                              const int* __restrict__ lengths,
                                              float* __restrict__ centers,
                                              float* __restrict__ rng,
                                              float* __restrict__ factor_out) {
  __shared__ float ld[SS];
  __shared__ float sA[SS];
  __shared__ float sB[SS];
  __shared__ float red[256];
  int b = blockIdx.x, tid = threadIdx.x;
#pragma unroll
  for (int j = 0; j < 2; ++j) {
    int s = tid + 256 * j;
    const float4* xr = (const float4*)(x2 + ((size_t)b * SS + s) * CIN);
    const float4* w0 = (const float4*)projw;
    const float4* w1 = (const float4*)(projw + CIN);
    float a0 = 0.f, a1 = 0.f;
    for (int i = 0; i < CIN / 4; ++i) {
      float4 xv = xr[i], v0 = w0[i], v1 = w1[i];
      a0 += xv.x * v0.x + xv.y * v0.y + xv.z * v0.z + xv.w * v0.w;
      a1 += xv.x * v1.x + xv.y * v1.y + xv.z * v1.z + xv.w * v1.w;
    }
    float p0 = a0 + projb[0], p1 = a1 + projb[1];
    ld[s] = p0;
    float m = mask[b * SS + s];
    float sp = fmaxf(p1, 0.f) + log1pf(__expf(-fabsf(p1)));  // softplus
    rng[b * SS + s] = sp * m;
  }
  __syncthreads();
  float mymax = -3.0e38f;
#pragma unroll
  for (int j = 0; j < 2; ++j) {
    int s = tid + 256 * j;
    if (mask[b * SS + s] > 0.f) mymax = fmaxf(mymax, ld[s]);
  }
  red[tid] = mymax;
  __syncthreads();
  for (int off = 128; off > 0; off >>= 1) {
    if (tid < off) red[tid] = fmaxf(red[tid], red[tid + off]);
    __syncthreads();
  }
  float mx = red[0];
  __syncthreads();
  float mysum = 0.f;
#pragma unroll
  for (int j = 0; j < 2; ++j) {
    int s = tid + 256 * j;
    if (mask[b * SS + s] > 0.f) mysum += __expf(ld[s] - mx);
  }
  red[tid] = mysum;
  __syncthreads();
  for (int off = 128; off > 0; off >>= 1) {
    if (tid < off) red[tid] += red[tid + off];
    __syncthreads();
  }
  float lse = mx + __logf(red[0]);
  float factor = __logf((float)lengths[b]) - lse;
  if (tid == 0) factor_out[b] = factor;
  __syncthreads();
#pragma unroll
  for (int j = 0; j < 2; ++j) {
    int s = tid + 256 * j;
    float d = __expf(ld[s] + factor) * mask[b * SS + s];
    ld[s] = d;
    sA[s] = d;
  }
  __syncthreads();
  float* src = sA;
  float* dst = sB;
  for (int off = 1; off < SS; off <<= 1) {  // Hillis-Steele inclusive scan
#pragma unroll
    for (int j = 0; j < 2; ++j) {
      int s = tid + 256 * j;
      dst[s] = src[s] + ((s >= off) ? src[s - off] : 0.f);
    }
    __syncthreads();
    float* tp = src; src = dst; dst = tp;
  }
#pragma unroll
  for (int j = 0; j < 2; ++j) {
    int s = tid + 256 * j;
    centers[b * SS + s] = src[s] - 0.5f * ld[s];
  }
}

// ---- Kernel 4: align row per (b,t), normalized + masked ----
__global__ __launch_bounds__(256) void k_align(const float* __restrict__ centers,
                                               const float* __restrict__ rng,
                                               const float* __restrict__ mask,
                                               const int* __restrict__ lengths,
                                               float* __restrict__ align) {
  __shared__ float av[SS];
  __shared__ float red[256];
  int b = blockIdx.x / TT, t = blockIdx.x % TT, tid = threadIdx.x;
  float acc = 0.f;
#pragma unroll
  for (int j = 0; j < 2; ++j) {
    int s = tid + 256 * j;
    float m = mask[b * SS + s];
    float r = rng[b * SS + s];
    float d = (float)t - centers[b * SS + s];
    float q = (r > 0.f) ? d / r : 0.f;
    float a = q * q;
    av[s] = a;
    acc += a * m;
  }
  red[tid] = acc;
  __syncthreads();
  for (int off = 128; off > 0; off >>= 1) {
    if (tid < off) red[tid] += red[tid + off];
    __syncthreads();
  }
  float denom = red[0] + 1e-5f;
  float melm = ((float)t < (float)lengths[b]) ? 1.f : 0.f;
  float* outrow = align + ((size_t)b * TT + t) * SS;
#pragma unroll
  for (int j = 0; j < 2; ++j) {
    int s = tid + 256 * j;
    outrow[s] = av[s] / denom * melm * mask[b * SS + s];
  }
}

// ---- Kernel 5: transpose inputs [B,S,C] -> [B,C,S] for contiguous B-fragments
__global__ __launch_bounds__(256) void k_transpose(const float* __restrict__ in,
                                                   float* __restrict__ inT) {
  int bs = blockIdx.x;
  int b = bs / SS, s = bs % SS, c = threadIdx.x;
  inT[(size_t)b * CIN * SS + (size_t)c * SS + s] =
      in[((size_t)b * SS + s) * CIN + c];
}

// ---- Kernel 6: ups[b] = align[b] @ inputs[b]  (M=2048, N=256, K=512) ----
// Each wave computes 16x64 (4 N-tiles): one A-fragment feeds 4 WMMAs.
__global__ __launch_bounds__(256) void k_upsample(const float* __restrict__ align,
                                                  const float* __restrict__ inT,
                                                  float* __restrict__ ups) {
  int wid = blockIdx.x * 8 + (threadIdx.x >> 5);  // 8192 waves
  int lane = threadIdx.x & 31, l16 = lane & 15, half = lane >> 4;
  int ng = wid & 3;            // group of 4 N-tiles
  int mt = (wid >> 2) & 127;
  int b = wid >> 9;
  int m0 = mt * 16;
  const float* A = align + (size_t)b * TT * SS;
  const float* Bm = inT + (size_t)b * CIN * SS;
  v8f c[4] = {};
  for (int ks = 0; ks < SS; ks += 32) {
    v16bf a = load_a_frag_g(A + (size_t)m0 * SS + ks, SS, l16, half);
#pragma unroll
    for (int j = 0; j < 4; ++j) {
      int n0 = (ng * 4 + j) * 16;
      v16bf bb = load_b_fragT(Bm + (size_t)n0 * SS + ks, SS, l16, half);
      c[j] = __builtin_amdgcn_wmma_f32_16x16x32_bf16(false, a, false, bb,
                                                     (short)0, c[j], false, false);
    }
  }
  float* out = ups + (size_t)b * TT * CIN;
#pragma unroll
  for (int j = 0; j < 4; ++j) {
    int n0 = (ng * 4 + j) * 16;
#pragma unroll
    for (int i = 0; i < 8; ++i) {
      int row = m0 + 8 * half + i;
      out[(size_t)row * CIN + n0 + l16] = c[j][i];
    }
  }
}

extern "C" void kernel_launch(void* const* d_in, const int* in_sizes, int n_in,
                              void* d_out, int out_size, void* d_ws,
                              size_t ws_size, hipStream_t stream) {
  const float* inputs = (const float*)d_in[0];
  const float* mask = (const float*)d_in[1];
  const int* lengths = (const int*)d_in[2];
  const float* W_ih = (const float*)d_in[3];
  const float* W_hh = (const float*)d_in[4];
  const float* b_ih = (const float*)d_in[5];
  const float* b_hh = (const float*)d_in[6];
  const float* proj_w = (const float*)d_in[7];
  const float* proj_b = (const float*)d_in[8];

  float* ups = (float*)d_out;                    // [B,T,C]
  float* align = ups + (size_t)BB * TT * CIN;    // [B,T,S]
  float* factor = align + (size_t)BB * TT * SS;  // [B]

  // Workspace: ~50.4 MB of f32 scratch
  float* ws = (float*)d_ws;
  float* xp = ws;                               // [2, B*S, 384]
  float* xA = xp + (size_t)2 * BB * SS * G3;    // [B*S, 256]
  float* xB = xA + (size_t)BB * SS * (2 * HH);  // [B*S, 256]
  float* ctr = xB + (size_t)BB * SS * (2 * HH); // [B,S]
  float* rng = ctr + BB * SS;                   // [B,S]
  float* inT = rng + BB * SS;                   // [B,C,S]

  const size_t lw = 2 * (size_t)G3 * CIN;  // W_ih layer stride
  const size_t lwh = 2 * (size_t)G3 * HH;  // W_hh layer stride
  const size_t lb = 2 * (size_t)G3;        // bias layer stride
  const size_t dynls = 2 * (size_t)16 * G3 * sizeof(float);  // gi double buffer

  // Layer 0
  k_xp_gemm<<<1024, 256, 0, stream>>>(inputs, W_ih, b_ih, xp);
  k_gru_scan<<<2, 256, dynls, stream>>>(xp, W_hh, b_hh, xA);
  // Layer 1
  k_xp_gemm<<<1024, 256, 0, stream>>>(xA, W_ih + lw, b_ih + lb, xp);
  k_gru_scan<<<2, 256, dynls, stream>>>(xp, W_hh + lwh, b_hh + lb, xB);
  // Head + upsampling
  k_head<<<BB, 256, 0, stream>>>(xB, proj_w, proj_b, mask, lengths, ctr, rng,
                                 factor);
  k_transpose<<<BB * SS, 256, 0, stream>>>(inputs, inT);
  k_align<<<BB * TT, 256, 0, stream>>>(ctr, rng, mask, lengths, align);
  k_upsample<<<1024, 256, 0, stream>>>(align, inT, ups);
}
